// FlowerAttention_62191126446224
// MI455X (gfx1250) — matrix operand order repaired and checked by reading, not compiled, not run on hardware
//
#include <hip/hip_runtime.h>

// ---------------------------------------------------------------------------
// MI455X (gfx1250, wave32) attention block:
//   qkv = x @ Wqkv^T  -> per-head RMSNorm + RoPE (+score scale folded into q)
//   flash attention (online softmax, f16 WMMA, f32 accum)
//   out = y @ Wproj^T
// All GEMM math on v_wmma_f32_16x16x32_f16. Double-buffered LDS pipelines:
// next K-slab global loads are issued between the barrier and the WMMA burst.
// ---------------------------------------------------------------------------

typedef __attribute__((ext_vector_type(16))) _Float16 v16h;
typedef __attribute__((ext_vector_type(8)))  _Float16 v8h;
typedef __attribute__((ext_vector_type(8)))  float    v8f;

#define BSZ      4
#define SEQ      2048
#define DIM      2048
#define NHEADS   16
#define HDIM     128
#define MROWS    (BSZ * SEQ)          // 8192
#define QKV_N    (3 * DIM)            // 6144

__device__ __forceinline__ v16h cat8(v8h lo, v8h hi) {
  return __builtin_shufflevector(lo, hi, 0,1,2,3,4,5,6,7,8,9,10,11,12,13,14,15);
}

__device__ __forceinline__ v8f wmma16(v16h a, v16h b, v8f c) {
  // D = A(16x32 f16) * B(32x16 f16) + C(16x16 f32)
  return __builtin_amdgcn_wmma_f32_16x16x32_f16(
      false, a, false, b, (short)0, c, false, false);
}

// A-fragment (16x32, MxK) from a row-major [16+][ld] f16 tile.
// lane: M = lane&15; elems 0..7 -> K = koff+0..7, elems 8..15 -> K = koff+16..23,
// koff = (lane&16)?8:0.  (ISA 7.12.2, 16-bit A 16x32 table)
__device__ __forceinline__ v16h frag_a(const _Float16* t, int ld, int lane) {
  const _Float16* p = t + (lane & 15) * ld + ((lane & 16) ? 8 : 0);
  v8h lo = *(const v8h*)(p);
  v8h hi = *(const v8h*)(p + 16);
  return cat8(lo, hi);
}

// B-fragment (32x16, KxN) from a tile stored [N][ld] (K contiguous per row).
// lane: N = lane&15; elems 0..7 -> K = koff+0..7, elems 8..15 -> K = koff+16..23, koff = (lane&16)?8:0.
__device__ __forceinline__ v16h frag_b(const _Float16* t, int ld, int lane) {
  const _Float16* p = t + (lane & 15) * ld + ((lane & 16) ? 8 : 0);
  v8h lo = *(const v8h*)(p);
  v8h hi = *(const v8h*)(p + 16);
  return cat8(lo, hi);
}

__device__ __forceinline__ v8h pack8(float4 a, float4 b) {
  return (v8h){(_Float16)a.x, (_Float16)a.y, (_Float16)a.z, (_Float16)a.w,
               (_Float16)b.x, (_Float16)b.y, (_Float16)b.z, (_Float16)b.w};
}

// ---------------------------------------------------------------------------
// Kernel 1/4: C[M,N] = A[M,K] * B[N,K]^T   (fp32 in, f16 WMMA, fp32 out)
// Block tile 128x128, 256 threads = 8 waves (2x4), wave tile 64x32.
// Double-buffered LDS, one barrier per K-step, next slab loads overlapped
// with the 8-WMMA burst.
// ---------------------------------------------------------------------------
__global__ __launch_bounds__(256) void gemm_abt(
    const float* __restrict__ A, const float* __restrict__ B,
    float* __restrict__ C, int M, int N, int K) {
  __shared__ __align__(16) _Float16 sA[2][128 * 32];
  __shared__ __align__(16) _Float16 sB[2][128 * 32];

  const int tid  = threadIdx.x;
  const int lane = tid & 31;
  const int wid  = tid >> 5;
  const int wm   = (wid >> 2) * 64;   // 0 or 64
  const int wn   = (wid & 3) * 32;    // 0,32,64,96
  const int bm   = blockIdx.y * 128;
  const int bn   = blockIdx.x * 128;

  const int row = tid >> 1;           // 0..127
  const int seg = (tid & 1) * 16;     // 0 or 16 (fp32 columns)
  const float* gA = A + (size_t)(bm + row) * K + seg;
  const float* gB = B + (size_t)(bn + row) * K + seg;
  const int wo = row * 32 + seg;      // LDS write offset (halves)

  v8f acc[4][2];
  for (int i = 0; i < 4; ++i)
    for (int j = 0; j < 2; ++j) acc[i][j] = {};

  // preload slab 0
  float4 ra[4], rb[4];
  for (int j = 0; j < 4; ++j) {
    ra[j] = *(const float4*)(gA + j * 4);
    rb[j] = *(const float4*)(gB + j * 4);
  }

  int p = 0;
  for (int k0 = 0; k0 < K; k0 += 32, p ^= 1) {
    // stage current slab (fp32 regs -> f16 LDS)
    *(v8h*)(sA[p] + wo)     = pack8(ra[0], ra[1]);
    *(v8h*)(sA[p] + wo + 8) = pack8(ra[2], ra[3]);
    *(v8h*)(sB[p] + wo)     = pack8(rb[0], rb[1]);
    *(v8h*)(sB[p] + wo + 8) = pack8(rb[2], rb[3]);
    __syncthreads();

    // issue next slab's global loads; latency hidden behind the WMMA burst
    if (k0 + 32 < K) {
      const float* nA = gA + k0 + 32;
      const float* nB = gB + k0 + 32;
      for (int j = 0; j < 4; ++j) {
        ra[j] = *(const float4*)(nA + j * 4);
        rb[j] = *(const float4*)(nB + j * 4);
      }
      __builtin_prefetch(nA + 32, 0, 1);   // global_prefetch_b8 (slab k0+64)
      __builtin_prefetch(nB + 32, 0, 1);
    }

    v16h af[4], bf[2];
    for (int mt = 0; mt < 4; ++mt)
      af[mt] = frag_a(sA[p] + (wm + mt * 16) * 32, 32, lane);
    for (int nt = 0; nt < 2; ++nt)
      bf[nt] = frag_b(sB[p] + (wn + nt * 16) * 32, 32, lane);
    for (int mt = 0; mt < 4; ++mt)
      for (int nt = 0; nt < 2; ++nt)
        acc[mt][nt] = wmma16(af[mt], bf[nt], acc[mt][nt]);
  }

  // C fragment: N = lane&15, M = r + ((lane&16)?8:0)
  const int roff = (lane & 16) ? 8 : 0;
  const int col  = lane & 15;
  for (int mt = 0; mt < 4; ++mt)
    for (int nt = 0; nt < 2; ++nt) {
      const int r0 = bm + wm + mt * 16 + roff;
      const int c  = bn + wn + nt * 16 + col;
      for (int r = 0; r < 8; ++r)
        C[(size_t)(r0 + r) * N + c] = acc[mt][nt][r];
    }
}

// ---------------------------------------------------------------------------
// Kernel 2/4: per-head RMSNorm + RoPE; q scaled by HDIM^-0.5; f16 out [b,h,s,d]
// One block = one (b,s,h) row of 128; 128 threads.
// ---------------------------------------------------------------------------
__global__ __launch_bounds__(128) void qkv_post(
    const float* __restrict__ QKV, const float* __restrict__ qw,
    const float* __restrict__ kw, _Float16* __restrict__ Qo,
    _Float16* __restrict__ Ko, _Float16* __restrict__ Vo) {
  const int s = blockIdx.x, h = blockIdx.y, b = blockIdx.z;
  const int d = threadIdx.x;
  const size_t base = ((size_t)(b * SEQ + s)) * QKV_N + h * HDIM + d;
  float q = QKV[base];
  float k = QKV[base + DIM];
  float v = QKV[base + 2 * DIM];

  __shared__ float sh[256 + 8];
  float sq = q * q, sk = k * k;
  for (int m = 16; m; m >>= 1) {
    sq += __shfl_xor(sq, m, 32);
    sk += __shfl_xor(sk, m, 32);
  }
  const int lane = d & 31, wid = d >> 5;
  if (lane == 0) { sh[256 + wid] = sq; sh[260 + wid] = sk; }
  __syncthreads();
  const float tq = sh[256] + sh[257] + sh[258] + sh[259];
  const float tk = sh[260] + sh[261] + sh[262] + sh[263];
  const float qn = q * rsqrtf(tq * (1.0f / HDIM) + 1e-6f) * qw[d];
  const float kn = k * rsqrtf(tk * (1.0f / HDIM) + 1e-6f) * kw[d];
  sh[d] = qn; sh[128 + d] = kn;
  __syncthreads();

  const int j = d & 63;
  const float ang = (float)s * __powf(32.0f, -(float)j * (1.0f / 64.0f));
  float sn, c;
  __sincosf(ang, &sn, &c);
  float qo, ko;
  if (d < 64) { qo = qn * c - sh[d + 64] * sn;       ko = kn * c - sh[128 + d + 64] * sn; }
  else        { qo = qn * c + sh[d - 64] * sn;       ko = kn * c + sh[128 + d - 64] * sn; }

  const size_t o = ((size_t)((b * NHEADS + h) * SEQ + s)) * HDIM + d;
  Qo[o] = (_Float16)(qo * 0.08838834764831845f);  // fold 1/sqrt(128) into q
  Ko[o] = (_Float16)ko;
  Vo[o] = (_Float16)v;
}

// ---------------------------------------------------------------------------
// Kernel 3/4: flash attention. Block = 128 q-rows of one (b,h); 8 waves,
// wave owns 16 q rows. Double-buffered K/V staging (64 keys/block, V
// transposed); next block's loads overlap the S/softmax/PV phase.
// ---------------------------------------------------------------------------
#define LDV 72   // padded key-stride of transposed V (rows 16B aligned)

__global__ __launch_bounds__(256) void flash_attn(
    const _Float16* __restrict__ Qg, const _Float16* __restrict__ Kg,
    const _Float16* __restrict__ Vg, float* __restrict__ Y) {
  __shared__ __align__(16) _Float16 sK[2][64 * HDIM];     // [key][dim]
  __shared__ __align__(16) _Float16 sV[2][HDIM * LDV];    // [dim][key] transposed
  __shared__ __align__(16) _Float16 sP[8 * 16 * 64];      // per-wave P tiles

  const int tid = threadIdx.x, lane = tid & 31, w = tid >> 5;
  const int qb = blockIdx.x * 128;
  const int bh = blockIdx.y;           // b*16 + h
  const int b  = bh >> 4, h = bh & 15;
  const size_t headOff = (size_t)bh * SEQ * HDIM;

  // Q fragments pinned in registers (A-side, 4 k-steps of 32 over D=128)
  const int qrow = qb + w * 16 + (lane & 15);
  const _Float16* qp = Qg + headOff + (size_t)qrow * HDIM + ((lane & 16) ? 8 : 0);
  v16h qf[4];
  for (int kk = 0; kk < 4; ++kk) {
    v8h lo = *(const v8h*)(qp + kk * 32);
    v8h hi = *(const v8h*)(qp + kk * 32 + 16);
    qf[kk] = cat8(lo, hi);
  }

  v8f o[8];
  for (int n = 0; n < 8; ++n) o[n] = {};
  float m_[8], l_[8];
  for (int r = 0; r < 8; ++r) { m_[r] = -3.0e38f; l_[r] = 0.0f; }
  _Float16* Pw = sP + w * 16 * 64;

  // staging thread mapping
  const int kr = tid >> 2;             // V: key row 0..63
  const int d0 = (tid & 3) * 32;       // V: dim chunk
  const _Float16* kbase = Kg + headOff;
  const _Float16* vbase = Vg + headOff;

  // preload block 0 into registers
  v8h kreg[4], vreg[4];
  for (int i = 0; i < 4; ++i)
    kreg[i] = ((const v8h*)kbase)[tid + i * 256];
  for (int jv = 0; jv < 4; ++jv)
    vreg[jv] = *(const v8h*)(vbase + (size_t)kr * HDIM + d0 + jv * 8);

  int p = 0;
  for (int kb = 0; kb < SEQ; kb += 64, p ^= 1) {
    // ---- stage current K/V block from registers ----
    {
      v8h* dst = (v8h*)sK[p];
      for (int i = 0; i < 4; ++i) dst[tid + i * 256] = kreg[i];
    }
    for (int jv = 0; jv < 32; ++jv)
      sV[p][(d0 + jv) * LDV + kr] = vreg[jv >> 3][jv & 7];
    __syncthreads();

    // ---- issue next block's global loads (hidden behind S/softmax/PV) ----
    if (kb + 64 < SEQ) {
      const v8h* nk = (const v8h*)(kbase + (size_t)(kb + 64) * HDIM);
      for (int i = 0; i < 4; ++i) kreg[i] = nk[tid + i * 256];
      const _Float16* nv = vbase + (size_t)(kb + 64 + kr) * HDIM + d0;
      for (int jv = 0; jv < 4; ++jv) vreg[jv] = *(const v8h*)(nv + jv * 8);
    }

    // ---- S = Q * K^T  (4 key-tiles of 16) ----
    v8f sacc[4];
    for (int n = 0; n < 4; ++n) sacc[n] = {};
    for (int kk = 0; kk < 4; ++kk)
      for (int n = 0; n < 4; ++n) {
        v16h bf = frag_b(sK[p] + (n * 16) * HDIM + kk * 32, HDIM, lane);
        sacc[n] = wmma16(qf[kk], bf, sacc[n]);
      }

    // ---- online softmax (rows live in 16-lane halves) ----
    float alpha[8];
    for (int r = 0; r < 8; ++r) {
      float cur = fmaxf(fmaxf(sacc[0][r], sacc[1][r]),
                        fmaxf(sacc[2][r], sacc[3][r]));
      for (int msk = 1; msk <= 8; msk <<= 1)
        cur = fmaxf(cur, __shfl_xor(cur, msk, 32));
      const float mn = fmaxf(m_[r], cur);
      alpha[r] = __expf(m_[r] - mn);
      m_[r] = mn;
    }
    float rs[8] = {0,0,0,0,0,0,0,0};
    for (int n = 0; n < 4; ++n)
      for (int r = 0; r < 8; ++r) {
        const float pe = __expf(sacc[n][r] - m_[r]);
        sacc[n][r] = pe;
        rs[r] += pe;
      }
    for (int r = 0; r < 8; ++r) {
      float t = rs[r];
      for (int msk = 1; msk <= 8; msk <<= 1) t += __shfl_xor(t, msk, 32);
      l_[r] = l_[r] * alpha[r] + t;
    }
    for (int n = 0; n < 8; ++n)
      for (int r = 0; r < 8; ++r) o[n][r] *= alpha[r];

    // ---- reshape P: C-layout regs -> row-major [16][64] f16 in LDS ----
    {
      const int roff = (lane & 16) ? 8 : 0;
      const int col  = lane & 15;
      for (int n = 0; n < 4; ++n)
        for (int r = 0; r < 8; ++r)
          Pw[(roff + r) * 64 + n * 16 + col] = (_Float16)sacc[n][r];
    }
    __syncthreads();

    // ---- O += P * V  (2 k-steps of 32 keys, 8 dim-tiles) ----
    for (int ks = 0; ks < 2; ++ks) {
      v16h pf = frag_a(Pw + ks * 32, 64, lane);
      for (int n = 0; n < 8; ++n) {
        v16h vf = frag_b(sV[p] + (n * 16) * LDV + ks * 32, LDV, lane);
        o[n] = wmma16(pf, vf, o[n]);
      }
    }
  }

  // ---- epilogue: O /= l, scatter to y[b, s, h*128 + d] ----
  const int roff = (lane & 16) ? 8 : 0;
  const int col  = lane & 15;
  for (int r = 0; r < 8; ++r) {
    const float inv = 1.0f / l_[r];
    const int srow = qb + w * 16 + roff + r;
    float* yp = Y + ((size_t)(b * SEQ + srow)) * DIM + h * HDIM + col;
    for (int n = 0; n < 8; ++n) yp[n * 16] = o[n][r] * inv;
  }
}

// ---------------------------------------------------------------------------
// Launch
// ---------------------------------------------------------------------------
extern "C" void kernel_launch(void* const* d_in, const int* in_sizes, int n_in,
                              void* d_out, int out_size, void* d_ws, size_t ws_size,
                              hipStream_t stream) {
  (void)in_sizes; (void)n_in; (void)out_size; (void)ws_size;
  const float* x     = (const float*)d_in[0];
  const float* wqkv  = (const float*)d_in[1];
  const float* wproj = (const float*)d_in[2];
  const float* qw    = (const float*)d_in[3];
  const float* kw    = (const float*)d_in[4];
  float* out = (float*)d_out;

  char* ws = (char*)d_ws;
  float*    qkv = (float*)ws;                                  // 8192*6144 f32
  _Float16* qh  = (_Float16*)(ws + (size_t)MROWS * QKV_N * 4); // 64*2048*128 f16
  _Float16* kh  = qh + (size_t)BSZ * NHEADS * SEQ * HDIM;
  _Float16* vh  = kh + (size_t)BSZ * NHEADS * SEQ * HDIM;
  float*    y   = (float*)(vh + (size_t)BSZ * NHEADS * SEQ * HDIM); // 8192*2048 f32

  // 1) qkv = x @ Wqkv^T
  gemm_abt<<<dim3(QKV_N / 128, MROWS / 128), 256, 0, stream>>>(
      x, wqkv, qkv, MROWS, QKV_N, DIM);
  // 2) RMSNorm + RoPE + split/scale -> f16 [b,h,s,d]
  qkv_post<<<dim3(SEQ, NHEADS, BSZ), 128, 0, stream>>>(qkv, qw, kw, qh, kh, vh);
  // 3) flash attention -> y [b,s,dim] f32
  flash_attn<<<dim3(SEQ / 128, BSZ * NHEADS), 256, 0, stream>>>(qh, kh, vh, y);
  // 4) out = y @ Wproj^T
  gemm_abt<<<dim3(DIM / 128, MROWS / 128), 256, 0, stream>>>(
      y, wproj, out, MROWS, DIM, DIM);
}
